// SpatialAttention_86517821216049
// MI455X (gfx1250) — matrix operand order repaired and checked by reading, not compiled, not run on hardware
//
#include <hip/hip_runtime.h>

#define B_N   32
#define IDF   256
#define CDF   768
#define SL    128
#define QL    4096      // 64*64
#define QTILE 128

typedef __attribute__((ext_vector_type(16))) __bf16 v16bf;
typedef __attribute__((ext_vector_type(8)))  __bf16 v8bf;
typedef __attribute__((ext_vector_type(8)))  float  v8f;

// ---------- helpers ----------
static __device__ __forceinline__ __bf16 f2bf(float f) {
  union { float f; unsigned u; } v; v.f = f;
  unsigned r = v.u + 0x7FFFu + ((v.u >> 16) & 1u);   // round-to-nearest-even
  unsigned short h = (unsigned short)(r >> 16);
  __bf16 b; __builtin_memcpy(&b, &h, 2);
  return b;
}

static __device__ __forceinline__ v8f wmma_bf16(v16bf a, v16bf b, v8f c) {
  // v_wmma_f32_16x16x32_bf16  D = A*B + C
  return __builtin_amdgcn_wmma_f32_16x16x32_bf16(false, a, false, b, (short)0, c, false, false);
}

static __device__ __forceinline__ v16bf join8(v8bf lo, v8bf hi) {
  return __builtin_shufflevector(lo, hi, 0,1,2,3,4,5,6,7,8,9,10,11,12,13,14,15);
}

// =====================================================================
// Stage 1: sourceT[b] = w_ctx (256x768) @ context[b] (768x128), f32 out.
// One block per batch, 8 waves, each wave owns a 32x128 output stripe.
// =====================================================================
#define KC      64
#define SA_STR  72   // 256 rows x 72 bf16 (144 B rows, 16B aligned)
#define SB_STR  72   // 128 rows x 72 bf16

__global__ __launch_bounds__(256)
void proj_ctx_kernel(const float* __restrict__ ctx, const float* __restrict__ w,
                     float* __restrict__ srcT) {
  __shared__ __bf16 sA[IDF * SA_STR];   // A:  [i][k]  (M=i, K=c)
  __shared__ __bf16 sB[SL  * SB_STR];   // B:  [s][k]  (N=s, K=c) transposed in LDS

  const int b    = blockIdx.x;
  const int tid  = threadIdx.x;
  const int lane = tid & 31, wave = tid >> 5;
  const int nl   = lane & 15, half = lane >> 4;

  v8f acc[2][8];
#pragma unroll
  for (int mt = 0; mt < 2; ++mt)
#pragma unroll
    for (int nt = 0; nt < 8; ++nt)
#pragma unroll
      for (int r = 0; r < 8; ++r) acc[mt][nt][r] = 0.f;

  for (int kc = 0; kc < CDF; kc += KC) {
    // w_ctx chunk: [256][64], global reads coalesced along c
    for (int e = tid; e < IDF * KC; e += 256) {
      int r = e >> 6, c = e & (KC - 1);
      sA[r * SA_STR + c] = f2bf(w[r * CDF + kc + c]);
    }
    // context chunk transposed: global coalesced along s, LDS [s][c]
    for (int e = tid; e < SL * KC; e += 256) {
      int s = e & (SL - 1), c = e >> 7;
      sB[s * SB_STR + c] = f2bf(ctx[((size_t)b * CDF + kc + c) * SL + s]);
    }
    __syncthreads();

#pragma unroll
    for (int ks = 0; ks < KC; ks += 32) {
      v16bf afrag[2];
#pragma unroll
      for (int mt = 0; mt < 2; ++mt) {
        const __bf16* pa = &sA[(wave * 32 + mt * 16 + nl) * SA_STR + ks];
        afrag[mt] = join8(*(const v8bf*)(pa + 8 * half),
                          *(const v8bf*)(pa + 16 + 8 * half));
      }
#pragma unroll
      for (int nt = 0; nt < 8; ++nt) {
        const __bf16* pb = &sB[(nt * 16 + nl) * SB_STR + ks];
        v16bf bfrag = join8(*(const v8bf*)(pb + 8 * half), *(const v8bf*)(pb + 16 + 8 * half));
#pragma unroll
        for (int mt = 0; mt < 2; ++mt)
          acc[mt][nt] = wmma_bf16(afrag[mt], bfrag, acc[mt][nt]);
      }
    }
    __syncthreads();
  }

  // write sourceT f32 to workspace: D vgpr r -> row r+8*half, lane%16 -> col
#pragma unroll
  for (int mt = 0; mt < 2; ++mt)
#pragma unroll
    for (int nt = 0; nt < 8; ++nt)
#pragma unroll
      for (int r = 0; r < 8; ++r) {
        int i = wave * 32 + mt * 16 + r + 8 * half;
        int s = nt * 16 + nl;
        srcT[((size_t)b * IDF + i) * SL + s] = acc[mt][nt][r];
      }
}

// =====================================================================
// Stage 2: fused attention per (batch, 128-query tile).
//   GEMM2  attn = target^T @ sourceT   (bf16 WMMA, f32 acc)
//   masked softmax (f32, reference's mask[q % B] tiling bug replicated)
//   GEMM3  wc = sourceT @ attn^T       (bf16 WMMA, f32 acc)
// LDS overlays, 200 KB of the 320 KB WGP pool.
// =====================================================================
#define TSTR 264   // bf16 row stride for [.][256] tiles (528 B, 16B aligned)
#define ISTR 136   // bf16 row stride for [.][128] tiles (272 B, 16B aligned)
#define ASTR 132   // f32  row stride for attn logits (528 B)

#define OFF_SIS  67584                       // after tileT/attnF region
#define OFF_SSI  (67584 + 69632)             // after Sis
#define SMEM2    (67584 + 69632 + 67584)     // 204800 B

__global__ __launch_bounds__(256)
void attn_kernel(const float* __restrict__ inp, const float* __restrict__ srcT,
                 const int* __restrict__ mask,
                 float* __restrict__ wc, float* __restrict__ attn_out) {
  extern __shared__ v8f smem_v[];
  char* smem = (char*)smem_v;
  __bf16* tileT = (__bf16*)smem;              // [128 q][264] : A of GEMM2 (target^T)
  float*  attnF = (float*)smem;               // [128 q][132] : overlays tileT
  __bf16* Sis   = (__bf16*)(smem + OFF_SIS);  // [256 i][136] : A of GEMM3
  __bf16* Ssi   = (__bf16*)(smem + OFF_SSI);  // [128 s][264] : B of GEMM2
  __bf16* attnB = (__bf16*)(smem + OFF_SSI);  // [128 q][136] : overlays Ssi (B of GEMM3)

  const int b     = blockIdx.y;
  const int qbase = blockIdx.x * QTILE;
  const int tid   = threadIdx.x;
  const int lane  = tid & 31, wave = tid >> 5;
  const int nl    = lane & 15, half = lane >> 4;

  // ---- stage LDS: sourceT in both layouts, input tile transposed ----
  for (int e = tid; e < IDF * SL; e += 256) {
    int s = e & (SL - 1), i = e >> 7;                       // coalesced along s
    __bf16 h = f2bf(srcT[((size_t)b * IDF + i) * SL + s]);
    Sis[i * ISTR + s] = h;
    Ssi[s * TSTR + i] = h;
  }
  for (int e = tid; e < IDF * QTILE; e += 256) {
    int q = e & (QTILE - 1), i = e >> 7;                    // coalesced along q
    tileT[q * TSTR + i] = f2bf(inp[((size_t)b * IDF + i) * QL + qbase + q]);
  }
  __syncthreads();

  // ---- GEMM2: each wave owns 16 query rows x 128 sources, K = 256 ----
  v8f acc[8];
#pragma unroll
  for (int nt = 0; nt < 8; ++nt)
#pragma unroll
    for (int r = 0; r < 8; ++r) acc[nt][r] = 0.f;

#pragma unroll
  for (int ks = 0; ks < IDF; ks += 32) {
    const __bf16* pa = &tileT[(wave * 16 + nl) * TSTR + ks];
    v16bf a = join8(*(const v8bf*)(pa + 8 * half),
                    *(const v8bf*)(pa + 16 + 8 * half));
#pragma unroll
    for (int nt = 0; nt < 8; ++nt) {
      const __bf16* pb = &Ssi[(nt * 16 + nl) * TSTR + ks];
      v16bf bb = join8(*(const v8bf*)(pb + 8 * half), *(const v8bf*)(pb + 16 + 8 * half));
      acc[nt] = wmma_bf16(a, bb, acc[nt]);
    }
  }
  __syncthreads();                 // all waves done reading tileT before overlay
#pragma unroll
  for (int nt = 0; nt < 8; ++nt)
#pragma unroll
    for (int r = 0; r < 8; ++r)
      attnF[(wave * 16 + r + 8 * half) * ASTR + nt * 16 + nl] = acc[nt][r];
  __syncthreads();

  // ---- masked softmax over s (row per thread); mask row = q_global % B ----
  if (tid < QTILE) {
    const int q  = tid;
    const int qg = qbase + q;
    const int* mr = mask + (qg & (B_N - 1)) * SL;
    float* rowp = &attnF[q * ASTR];
    float mx = -3.4e38f;
    for (int s = 0; s < SL; ++s) if (!mr[s]) mx = fmaxf(mx, rowp[s]);
    float sum = 0.f;
    for (int s = 0; s < SL; ++s) {
      float e = mr[s] ? 0.f : __expf(rowp[s] - mx);
      rowp[s] = e; sum += e;
    }
    float inv = 1.0f / sum;
    for (int s = 0; s < SL; ++s) {
      float p = rowp[s] * inv;
      rowp[s] = p;
      attnB[q * ISTR + s] = f2bf(p);
      attn_out[((size_t)b * SL + s) * QL + qg] = p;   // transposed write, coalesced in q
    }
  }
  __syncthreads();

  // ---- GEMM3: each wave owns 32 channel rows x 128 queries, K = 128 ----
  v8f acc2[2][8];
#pragma unroll
  for (int mt = 0; mt < 2; ++mt)
#pragma unroll
    for (int nt = 0; nt < 8; ++nt)
#pragma unroll
      for (int r = 0; r < 8; ++r) acc2[mt][nt][r] = 0.f;

#pragma unroll
  for (int ks = 0; ks < SL; ks += 32) {
    v16bf a2[2];
#pragma unroll
    for (int mt = 0; mt < 2; ++mt) {
      const __bf16* pa = &Sis[(wave * 32 + mt * 16 + nl) * ISTR + ks];
      a2[mt] = join8(*(const v8bf*)(pa + 8 * half),
                     *(const v8bf*)(pa + 16 + 8 * half));
    }
#pragma unroll
    for (int nt = 0; nt < 8; ++nt) {
      const __bf16* pb = &attnB[(nt * 16 + nl) * ISTR + ks];
      v16bf bb = join8(*(const v8bf*)(pb + 8 * half), *(const v8bf*)(pb + 16 + 8 * half));
#pragma unroll
      for (int mt = 0; mt < 2; ++mt)
        acc2[mt][nt] = wmma_bf16(a2[mt], bb, acc2[mt][nt]);
    }
  }

#pragma unroll
  for (int mt = 0; mt < 2; ++mt)
#pragma unroll
    for (int nt = 0; nt < 8; ++nt)
#pragma unroll
      for (int r = 0; r < 8; ++r) {
        int i = wave * 32 + mt * 16 + r + 8 * half;
        int q = qbase + nt * 16 + nl;
        wc[((size_t)b * IDF + i) * QL + q] = acc2[mt][nt][r];
      }
}

// =====================================================================
extern "C" void kernel_launch(void* const* d_in, const int* in_sizes, int n_in,
                              void* d_out, int out_size, void* d_ws, size_t ws_size,
                              hipStream_t stream) {
  (void)in_sizes; (void)n_in; (void)out_size; (void)ws_size;
  const float* inp  = (const float*)d_in[0];   // [B, IDF, IH, IW]
  const float* ctx  = (const float*)d_in[1];   // [B, CDF, SL]
  const float* w    = (const float*)d_in[2];   // [IDF, CDF]
  const int*   mask = (const int*)d_in[3];     // [B, SL]

  float* wc       = (float*)d_out;                         // [B, IDF, QL]
  float* attn_out = wc + (size_t)B_N * IDF * QL;           // [B, SL, QL]
  float* srcT     = (float*)d_ws;                          // [B, IDF, SL] f32 (4 MB)

  // 200 KB dynamic LDS (WGP has 320 KB); raise the runtime cap.
  hipFuncSetAttribute(reinterpret_cast<const void*>(attn_kernel),
                      hipFuncAttributeMaxDynamicSharedMemorySize, SMEM2);

  proj_ctx_kernel<<<B_N, 256, 0, stream>>>(ctx, w, srcT);
  attn_kernel<<<dim3(QL / QTILE, B_N), 256, SMEM2, stream>>>(inp, srcT, mask, wc, attn_out);
}
